// FlowPolicy_45964740002289
// MI455X (gfx1250) — matrix operand order repaired and checked by reading, not compiled, not run on hardware
//
#include <hip/hip_runtime.h>
#include <hip/hip_bf16.h>

// ---------------------------------------------------------------------------
// FlowPolicy fused pipeline for gfx1250 (MI455X).
// Precompute: weight-norm flow weights (f16) + logdet diag blocks (f32),
//             actor weights transposed to [n][k] f16.
// Main: per-WG 32 rows, fully fused actor MLP + flows using
//       v_wmma_f32_16x16x32_f16, activations staged in LDS.
// Round 2: branch-free templated epilogue + dual accumulator chains per
//          tile-pair for WMMA ILP.
// ---------------------------------------------------------------------------

#define BATCH   65536
#define OBSD    256
#define HID     256
#define DD      64
#define MT      32          // rows per workgroup
#define NTHR    128         // 4 waves
#define XSTR    264         // f16 LDS row stride (halves)
#define YSTR    264         // f32 LDS row stride (floats)

#define LN2C    0.6931471805599453f
#define LOG2PIC 1.8378770664093453f

typedef __attribute__((ext_vector_type(16))) _Float16 v16h;
typedef __attribute__((ext_vector_type(8)))  float    v8f;

union FragU { v16h v; uint4 q[2]; };

// Epilogue modes
#define EP_RELU_F16  0
#define EP_F32       1

// ---------------- precompute kernels ----------------

// out[n*K + k] = (f16) in[k*N + n]
__global__ void k_transpose_f16(const float* __restrict__ in, _Float16* __restrict__ out,
                                int K, int N) {
    int idx = blockIdx.x * blockDim.x + threadIdx.x;
    if (idx >= K * N) return;
    int n = idx / K, k = idx % K;
    out[n * K + k] = (_Float16)in[k * N + n];
}

// Weight-norm masked linear prep. One block (64 thr) per output row o.
// v = exp(W)*md + W*mo ; w = exp(logg)*v/||v|| (f16, natural [o][i] layout)
// ld[o*in_pd + j] = logg[o] + W[o][d*in_pd+j] - log||v||   (diag block only)
__global__ void k_flowprep(const float* __restrict__ W, const float* __restrict__ logg,
                           _Float16* __restrict__ wOut, float* __restrict__ ldOut,
                           int inF, int outF) {
    int o = blockIdx.x;
    int t = threadIdx.x;                    // 64 threads
    int out_pd = outF >> 6, in_pd = inF >> 6;
    int d   = o / out_pd;
    int dlo = d * in_pd, dhi = dlo + in_pd;
    __shared__ float red[64];
    float partial = 0.f;
    for (int i = t; i < inF; i += 64) {
        float wv = W[o * inF + i];
        float v  = (i >= dlo && i < dhi) ? __expf(wv) : ((i < dlo) ? wv : 0.f);
        partial += v * v;
    }
    red[t] = partial;
    __syncthreads();
    for (int s = 32; s > 0; s >>= 1) {
        if (t < s) red[t] += red[t + s];
        __syncthreads();
    }
    float vnorm = sqrtf(red[0]);
    float scale = __expf(logg[o]) / vnorm;
    float logvn = logf(vnorm);
    for (int i = t; i < inF; i += 64) {
        float wv = W[o * inF + i];
        float v  = (i >= dlo && i < dhi) ? __expf(wv) : ((i < dlo) ? wv : 0.f);
        wOut[o * inF + i] = (_Float16)(v * scale);
        if (i >= dlo && i < dhi)
            ldOut[o * in_pd + (i - dlo)] = logg[o] + wv - logvn;
    }
}

// ---------------- fragment loaders (CDNA5 16-bit WMMA layouts) ----------------

// A 16x32 f16 from row-major LDS: lanes 0-15 row=lane, K {0..7,16..23};
// lanes 16-31 same rows, K {8..15,24..31}.
__device__ __forceinline__ v16h load_a_frag(const _Float16* lds, int rowBase,
                                            int kBase, int lane) {
    int row  = rowBase + (lane & 15);
    int kOff = kBase + ((lane & 16) ? 8 : 0);
    const _Float16* p = lds + row * XSTR + kOff;
    FragU u;
    u.q[0] = *(const uint4*)(p);        // K kOff..kOff+7
    u.q[1] = *(const uint4*)(p + 16);   // K kOff+16..kOff+23
    return u.v;
}

// B 32x16 f16 from global weights stored [n][k] (ldb = inF):
// lane n = lane%16, K {0..7,16..23} (lanes 0-15) / {8..15,24..31} (lanes 16-31).
__device__ __forceinline__ v16h load_b_frag(const _Float16* __restrict__ w, int ldb,
                                            int nBase, int kBase, int lane) {
    int n    = nBase + (lane & 15);
    int kOff = kBase + ((lane & 16) ? 8 : 0);
    const _Float16* p = w + n * ldb + kOff;
    FragU u;
    u.q[0] = *(const uint4*)(p);
    u.q[1] = *(const uint4*)(p + 16);
    return u.v;
}

// Branch-free epilogue: bias add, then either relu->f16 into LDS xbuf or raw
// f32 into ybuf.  C/D fragment mapping: m = v + 8*(lane>=16), n = lane%16.
template<int MODE>
__device__ __forceinline__ void epilogue(v8f acc, int nBase, int mW, int lane,
                                         const float* __restrict__ bias,
                                         float* yOut, _Float16* dstLds) {
    int n = nBase + (lane & 15);
    float bv = bias[n];
    int rbase = mW + ((lane & 16) ? 8 : 0);
#pragma unroll
    for (int v = 0; v < 8; ++v) {
        float val = acc[v] + bv;
        int row = rbase + v;
        if constexpr (MODE == EP_RELU_F16) {
            dstLds[row * XSTR + n] = (_Float16)fmaxf(val, 0.f);
        } else {
            yOut[row * YSTR + n] = val;
        }
    }
}

// One GEMM stage for a wave: rows [mW, mW+16), cols [colBase, colBase+16*nTiles),
// K = 32*KT from LDS activations.  Tiles processed in pairs with two
// independent accumulator chains for WMMA ILP.  nTiles must be even.
template<int KT, int MODE>
__device__ __forceinline__ void gemm_stage(
    const _Float16* aLds, const _Float16* __restrict__ w, int ldb,
    int colBase, int nTiles, int mW, int lane,
    const float* __restrict__ bias,
    float* yOut, _Float16* dstLds)
{
    v16h afr[KT];
#pragma unroll
    for (int kt = 0; kt < KT; ++kt)
        afr[kt] = load_a_frag(aLds, mW, kt * 32, lane);
    for (int t = 0; t < nTiles; t += 2) {
        int n0 = colBase + t * 16;
        int n1 = n0 + 16;
        v8f acc0 = {};
        v8f acc1 = {};
#pragma unroll
        for (int kt = 0; kt < KT; ++kt) {
            v16h b0 = load_b_frag(w, ldb, n0, kt * 32, lane);
            v16h b1 = load_b_frag(w, ldb, n1, kt * 32, lane);
            acc0 = __builtin_amdgcn_wmma_f32_16x16x32_f16(
                false, afr[kt], false, b0, (short)0, acc0, false, false);
            acc1 = __builtin_amdgcn_wmma_f32_16x16x32_f16(
                false, afr[kt], false, b1, (short)0, acc1, false, false);
        }
        epilogue<MODE>(acc0, n0, mW, lane, bias, yOut, dstLds);
        epilogue<MODE>(acc1, n1, mW, lane, bias, yOut, dstLds);
    }
}

// Flow elementwise pass: logsumexp logdet recurrence + tanh; per (m,d) pair.
template<int OPD, int IPD, bool FINAL>
__device__ __forceinline__ void flow_pass2(
    const float* ybuf, float* sldb, const float* __restrict__ ldl,
    _Float16* dst, int tid, int wgBase, float* __restrict__ outX)
{
    for (int idx = tid; idx < MT * DD; idx += NTHR) {
        int m = idx >> 6, d = idx & 63;
        float so[IPD];
#pragma unroll
        for (int j = 0; j < IPD; ++j) so[j] = sldb[(m * DD + d) * 4 + j];
        float sn[OPD];
        float lastT = 0.f;
#pragma unroll
        for (int i = 0; i < OPD; ++i) {
            float lse;
            if constexpr (IPD == 1) {
                lse = so[0] + ldl[(d * OPD + i)];
            } else {
                float tmp[IPD];
                float mx = -1e30f;
#pragma unroll
                for (int j = 0; j < IPD; ++j) {
                    tmp[j] = so[j] + ldl[(d * OPD + i) * IPD + j];
                    mx = fmaxf(mx, tmp[j]);
                }
                float s = 0.f;
#pragma unroll
                for (int j = 0; j < IPD; ++j) s += __expf(tmp[j] - mx);
                lse = mx + logf(s);
            }
            float y  = ybuf[m * YSTR + d * OPD + i];
            float z  = -2.f * y;
            float sp = (z > 0.f) ? (z + log1pf(__expf(-z))) : log1pf(__expf(z));
            float tld = -2.f * (y - LN2C + sp);
            float tv  = tanhf(y);
            if constexpr (!FINAL) dst[m * XSTR + d * OPD + i] = (_Float16)tv;
            lastT = tv;
            sn[i] = lse + tld;
        }
        if constexpr (FINAL) {
            outX[(size_t)(wgBase + m) * DD + d] = lastT;
            // stash per-(m,d) logprob contribution for the reduction pass
            sldb[(m * DD + d) * 4] = -0.5f * lastT * lastT - 0.5f * LOG2PIC - sn[0];
        } else {
#pragma unroll
            for (int i = 0; i < OPD; ++i) sldb[(m * DD + d) * 4 + i] = sn[i];
        }
    }
}

// ---------------- main fused kernel ----------------

__global__ __launch_bounds__(NTHR) void k_flowpolicy_main(
    const float* __restrict__ obs, const float* __restrict__ raw,
    const float* __restrict__ ab0, const float* __restrict__ ab1,
    const float* __restrict__ ab2,
    const float* __restrict__ fb0, const float* __restrict__ fb1,
    const float* __restrict__ fb2, const float* __restrict__ fb3,
    const _Float16* __restrict__ aW0T, const _Float16* __restrict__ aW1T,
    const _Float16* __restrict__ aW2T,
    const _Float16* __restrict__ fw0, const _Float16* __restrict__ fw1,
    const _Float16* __restrict__ fw2, const _Float16* __restrict__ fw3,
    const float* __restrict__ ld0, const float* __restrict__ ld1,
    const float* __restrict__ ld2, const float* __restrict__ ld3,
    float* __restrict__ outX, float* __restrict__ outLP)
{
    extern __shared__ char smem[];
    _Float16* xA   = (_Float16*)smem;                    // MT x XSTR f16
    _Float16* xB   = xA + MT * XSTR;                     // MT x XSTR f16
    float*    ybuf = (float*)(xB + MT * XSTR);           // MT x YSTR f32
    float*    sldb = ybuf + MT * YSTR;                   // MT x 64 x 4 f32

    int tid   = threadIdx.x;
    int lane  = tid & 31;
    int wave  = tid >> 5;
    int waveM = wave & 1;     // row half
    int waveN = wave >> 1;    // col half
    int mW    = waveM << 4;
    int wgBase = blockIdx.x * MT;

    // S0: obs tile -> f16 LDS (vectorized float4)
    const float4* obs4 = (const float4*)(obs + (size_t)wgBase * OBSD);
    for (int idx = tid; idx < MT * (OBSD / 4); idx += NTHR) {
        int m = idx >> 6, c4 = idx & 63;
        float4 o = obs4[m * 64 + c4];
        _Float16* dst = xA + m * XSTR + c4 * 4;
        dst[0] = (_Float16)o.x; dst[1] = (_Float16)o.y;
        dst[2] = (_Float16)o.z; dst[3] = (_Float16)o.w;
    }
    __syncthreads();

    // Actor MLP
    gemm_stage<8, EP_RELU_F16>(xA, aW0T, 256, waveN * 128, 8, mW, lane, ab0, nullptr, xB);
    __syncthreads();
    gemm_stage<8, EP_RELU_F16>(xB, aW1T, 256, waveN * 128, 8, mW, lane, ab1, nullptr, xA);
    __syncthreads();
    gemm_stage<8, EP_F32>(xA, aW2T, 256, waveN * 64, 4, mW, lane, ab2, ybuf, nullptr);
    __syncthreads();

    // Reparameterized sample; seed sld with clipped logstd (constant offset
    // commutes through logsumexp -> equals adding logstd to final sum_logdet)
    for (int idx = tid; idx < MT * DD; idx += NTHR) {
        int m = idx >> 6, d = idx & 63;
        float mu = ybuf[m * YSTR + d];
        float ls = fminf(fmaxf(ybuf[m * YSTR + 64 + d], -20.f), 2.f);
        float rv = raw[(size_t)(wgBase + m) * DD + d];
        float xv = mu + __expf(ls) * rv;
        xA[m * XSTR + d] = (_Float16)xv;
        sldb[(m * DD + d) * 4] = ls;
    }
    __syncthreads();

    // Flow layer 0: 64 -> 256
    gemm_stage<2, EP_F32>(xA, fw0, 64, waveN * 128, 8, mW, lane, fb0, ybuf, nullptr);
    __syncthreads();
    flow_pass2<4, 1, false>(ybuf, sldb, ld0, xB, tid, wgBase, outX);
    __syncthreads();
    // Flow layer 1: 256 -> 256
    gemm_stage<8, EP_F32>(xB, fw1, 256, waveN * 128, 8, mW, lane, fb1, ybuf, nullptr);
    __syncthreads();
    flow_pass2<4, 4, false>(ybuf, sldb, ld1, xA, tid, wgBase, outX);
    __syncthreads();
    // Flow layer 2: 256 -> 256
    gemm_stage<8, EP_F32>(xA, fw2, 256, waveN * 128, 8, mW, lane, fb2, ybuf, nullptr);
    __syncthreads();
    flow_pass2<4, 4, false>(ybuf, sldb, ld2, xB, tid, wgBase, outX);
    __syncthreads();
    // Flow layer 3: 256 -> 64 (final: write x, stash logprob contributions)
    gemm_stage<8, EP_F32>(xB, fw3, 256, waveN * 32, 2, mW, lane, fb3, ybuf, nullptr);
    __syncthreads();
    flow_pass2<1, 4, true>(ybuf, sldb, ld3, nullptr, tid, wgBase, outX);
    __syncthreads();

    // logprob reduction over d
    if (tid < MT) {
        float s = 0.f;
        for (int d = 0; d < DD; ++d) s += sldb[(tid * DD + d) * 4];
        outLP[wgBase + tid] = s;
    }
}

// ---------------- host launcher ----------------

extern "C" void kernel_launch(void* const* d_in, const int* in_sizes, int n_in,
                              void* d_out, int out_size, void* d_ws, size_t ws_size,
                              hipStream_t stream) {
    const float* obs  = (const float*)d_in[0];
    const float* raw  = (const float*)d_in[1];
    const float* aW0  = (const float*)d_in[2];
    const float* ab0  = (const float*)d_in[3];
    const float* aW1  = (const float*)d_in[4];
    const float* ab1  = (const float*)d_in[5];
    const float* aW2  = (const float*)d_in[6];
    const float* ab2  = (const float*)d_in[7];
    const float* fW0  = (const float*)d_in[8];
    const float* fg0  = (const float*)d_in[9];
    const float* fb0  = (const float*)d_in[10];
    const float* fW1  = (const float*)d_in[11];
    const float* fg1  = (const float*)d_in[12];
    const float* fb1  = (const float*)d_in[13];
    const float* fW2  = (const float*)d_in[14];
    const float* fg2  = (const float*)d_in[15];
    const float* fb2  = (const float*)d_in[16];
    const float* fW3  = (const float*)d_in[17];
    const float* fg3  = (const float*)d_in[18];
    const float* fb3  = (const float*)d_in[19];

    // workspace layout (bytes), all 16B aligned
    char* ws = (char*)d_ws;
    _Float16* aW0T = (_Float16*)(ws + 0);        // 256*256 f16 = 131072 B
    _Float16* aW1T = (_Float16*)(ws + 131072);   // 131072 B
    _Float16* aW2T = (_Float16*)(ws + 262144);   // 128*256 f16 = 65536 B
    _Float16* fw0  = (_Float16*)(ws + 327680);   // 256*64  f16 = 32768 B
    _Float16* fw1  = (_Float16*)(ws + 360448);   // 131072 B
    _Float16* fw2  = (_Float16*)(ws + 491520);   // 131072 B
    _Float16* fw3  = (_Float16*)(ws + 622592);   // 64*256 f16 = 32768 B
    float*    ld0  = (float*)(ws + 655360);      // 256*1 f32
    float*    ld1  = (float*)(ws + 656384);      // 256*4 f32
    float*    ld2  = (float*)(ws + 660480);      // 256*4 f32
    float*    ld3  = (float*)(ws + 664576);      // 64*4  f32

    float* outX  = (float*)d_out;                // (1,B,64) = 4194304 floats
    float* outLP = outX + (size_t)BATCH * DD;    // (1,B,1)  = 65536 floats

    // precompute: weight transposes (f16) + flow weight-norm / logdet consts
    k_transpose_f16<<<256, 256, 0, stream>>>(aW0, aW0T, 256, 256);
    k_transpose_f16<<<256, 256, 0, stream>>>(aW1, aW1T, 256, 256);
    k_transpose_f16<<<128, 256, 0, stream>>>(aW2, aW2T, 256, 128);
    k_flowprep<<<256, 64, 0, stream>>>(fW0, fg0, fw0, ld0, 64, 256);
    k_flowprep<<<256, 64, 0, stream>>>(fW1, fg1, fw1, ld1, 256, 256);
    k_flowprep<<<256, 64, 0, stream>>>(fW2, fg2, fw2, ld2, 256, 256);
    k_flowprep<<<64, 64, 0, stream>>>(fW3, fg3, fw3, ld3, 256, 64);

    size_t shmem = (size_t)(2 * MT * XSTR) * sizeof(_Float16)   // xA, xB
                 + (size_t)(MT * YSTR) * sizeof(float)          // ybuf
                 + (size_t)(MT * DD * 4) * sizeof(float);       // sld
    k_flowpolicy_main<<<BATCH / MT, NTHR, shmem, stream>>>(
        obs, raw, ab0, ab1, ab2, fb0, fb1, fb2, fb3,
        aW0T, aW1T, aW2T, fw0, fw1, fw2, fw3,
        ld0, ld1, ld2, ld3, outX, outLP);
}
